// MSDeformRefAttn2D_90074054132154
// MI455X (gfx1250) — matrix operand order repaired and checked by reading, not compiled, not run on hardware
//
#include <hip/hip_runtime.h>
#include <hip/hip_bf16.h>
#include <math.h>

// ---------------------------------------------------------------------------
// MSDeformRefAttn2D for MI455X (gfx1250): bf16 WMMA GEMMs + L2-resident
// deformable bilinear sampling.
// B=1, Ce=256, Hq=Wq=100 (Q=10000), Nc=L=6, Hl=32, Wl=88, NH=8, NP=4, Dh=32
// ---------------------------------------------------------------------------

#define QTOT   10000
#define CE     256
#define NCAM   6
#define HL     32
#define WL     88
#define NHEAD  8
#define NPNT   4
#define DH     32

typedef __bf16 bf16_t;
typedef __attribute__((ext_vector_type(16))) bf16_t v16bf;
typedef __attribute__((ext_vector_type(8)))  float  v8f;

union FragAB { v16bf v; unsigned int d[8]; };
union FragC  { v8f   v; float f[8]; };

__device__ inline unsigned short f2bf(float f) {
  unsigned int u = __float_as_uint(f);
  unsigned int r = u + 0x7FFFu + ((u >> 16) & 1u);   // round-to-nearest-even
  return (unsigned short)(r >> 16);
}

// ---------------------------------------------------------------------------
// 1) Convert the four weight matrices fp32 -> bf16
// ---------------------------------------------------------------------------
__global__ void cvt_weights_kernel(const float* __restrict__ w_in,
                                   const float* __restrict__ w_off,
                                   const float* __restrict__ w_att,
                                   const float* __restrict__ w_out,
                                   unsigned short* __restrict__ b_in,
                                   unsigned short* __restrict__ b_off,
                                   unsigned short* __restrict__ b_att,
                                   unsigned short* __restrict__ b_out) {
  int i = blockIdx.x * blockDim.x + threadIdx.x;
  if (i < CE * CE)          b_in[i]  = f2bf(w_in[i]);
  if (i < NHEAD*NPNT*2*CE)  b_off[i] = f2bf(w_off[i]);
  if (i < NHEAD*NPNT*CE)    b_att[i] = f2bf(w_att[i]);
  if (i < CE * CE)          b_out[i] = f2bf(w_out[i]);
}

// ---------------------------------------------------------------------------
// 2) 4x4 inverse (adjugate) + BEV grid projection -> base uv + visibility
// ---------------------------------------------------------------------------
__device__ inline void inv4(const float* m, float* o) {
  float inv[16];
  inv[0]  =  m[5]*m[10]*m[15] - m[5]*m[11]*m[14] - m[9]*m[6]*m[15] + m[9]*m[7]*m[14] + m[13]*m[6]*m[11] - m[13]*m[7]*m[10];
  inv[4]  = -m[4]*m[10]*m[15] + m[4]*m[11]*m[14] + m[8]*m[6]*m[15] - m[8]*m[7]*m[14] - m[12]*m[6]*m[11] + m[12]*m[7]*m[10];
  inv[8]  =  m[4]*m[9]*m[15]  - m[4]*m[11]*m[13] - m[8]*m[5]*m[15] + m[8]*m[7]*m[13] + m[12]*m[5]*m[11] - m[12]*m[7]*m[9];
  inv[12] = -m[4]*m[9]*m[14]  + m[4]*m[10]*m[13] + m[8]*m[5]*m[14] - m[8]*m[6]*m[13] - m[12]*m[5]*m[10] + m[12]*m[6]*m[9];
  inv[1]  = -m[1]*m[10]*m[15] + m[1]*m[11]*m[14] + m[9]*m[2]*m[15] - m[9]*m[3]*m[14] - m[13]*m[2]*m[11] + m[13]*m[3]*m[10];
  inv[5]  =  m[0]*m[10]*m[15] - m[0]*m[11]*m[14] - m[8]*m[2]*m[15] + m[8]*m[3]*m[14] + m[12]*m[2]*m[11] - m[12]*m[3]*m[10];
  inv[9]  = -m[0]*m[9]*m[15]  + m[0]*m[11]*m[13] + m[8]*m[1]*m[15] - m[8]*m[3]*m[13] - m[12]*m[1]*m[11] + m[12]*m[3]*m[9];
  inv[13] =  m[0]*m[9]*m[14]  - m[0]*m[10]*m[13] - m[8]*m[1]*m[14] + m[8]*m[2]*m[13] + m[12]*m[1]*m[10] - m[12]*m[2]*m[9];
  inv[2]  =  m[1]*m[6]*m[15]  - m[1]*m[7]*m[14]  - m[5]*m[2]*m[15] + m[5]*m[3]*m[14] + m[13]*m[2]*m[7]  - m[13]*m[3]*m[6];
  inv[6]  = -m[0]*m[6]*m[15]  + m[0]*m[7]*m[14]  + m[4]*m[2]*m[15] - m[4]*m[3]*m[14] - m[12]*m[2]*m[7]  + m[12]*m[3]*m[6];
  inv[10] =  m[0]*m[5]*m[15]  - m[0]*m[7]*m[13]  - m[4]*m[1]*m[15] + m[4]*m[3]*m[13] + m[12]*m[1]*m[7]  - m[12]*m[3]*m[5];
  inv[14] = -m[0]*m[5]*m[14]  + m[0]*m[6]*m[13]  + m[4]*m[1]*m[14] - m[4]*m[2]*m[13] - m[12]*m[1]*m[6]  + m[12]*m[2]*m[5];
  inv[3]  = -m[1]*m[6]*m[11]  + m[1]*m[7]*m[10]  + m[5]*m[2]*m[11] - m[5]*m[3]*m[10] - m[9]*m[2]*m[7]   + m[9]*m[3]*m[6];
  inv[7]  =  m[0]*m[6]*m[11]  - m[0]*m[7]*m[10]  - m[4]*m[2]*m[11] + m[4]*m[3]*m[10] + m[8]*m[2]*m[7]   - m[8]*m[3]*m[6];
  inv[11] = -m[0]*m[5]*m[11]  + m[0]*m[7]*m[9]   + m[4]*m[1]*m[11] - m[4]*m[3]*m[9]  - m[8]*m[1]*m[7]   + m[8]*m[3]*m[5];
  inv[15] =  m[0]*m[5]*m[10]  - m[0]*m[6]*m[9]   - m[4]*m[1]*m[10] + m[4]*m[2]*m[9]  + m[8]*m[1]*m[6]   - m[8]*m[2]*m[5];
  float det = m[0]*inv[0] + m[1]*inv[4] + m[2]*inv[8] + m[3]*inv[12];
  float id = 1.0f / det;
  for (int i = 0; i < 16; ++i) o[i] = inv[i] * id;
}

__global__ __launch_bounds__(256)
void proj_kernel(const float* __restrict__ img2lidar,  // [Nc,4,4]
                 const float* __restrict__ img_wh,     // [Nc,2]
                 float* __restrict__ bx, float* __restrict__ by,
                 float* __restrict__ vis) {
  __shared__ float s_inv[16];
  int cam = blockIdx.y;
  if (threadIdx.x == 0) {
    float mm[16];
    for (int i = 0; i < 16; ++i) {
      float v = img2lidar[cam * 16 + i];
      if (isnan(v)) v = 0.0f;
      else if (isinf(v)) v = (v > 0.0f) ? 1e6f : -1e6f;
      mm[i] = v;
    }
    inv4(mm, s_inv);
  }
  __syncthreads();
  int q = blockIdx.x * 256 + threadIdx.x;
  if (q >= QTOT) return;
  int yq = q / 100, xq = q % 100;
  float X = ((float)xq + 0.5f) * 0.01f * 102.4f - 51.2f;
  float Y = ((float)yq + 0.5f) * 0.01f * 102.4f - 51.2f;
  float Z = -1.0f;  // 0.5*(3-(-5)) + (-5)
  float c0 = s_inv[0] * X + s_inv[1] * Y + s_inv[2]  * Z + s_inv[3];
  float c1 = s_inv[4] * X + s_inv[5] * Y + s_inv[6]  * Z + s_inv[7];
  float c2 = s_inv[8] * X + s_inv[9] * Y + s_inv[10] * Z + s_inv[11];
  float depth = c2;
  float dz = fmaxf(depth, 1e-5f);
  float un = (c0 / dz) / img_wh[cam * 2 + 0];
  float vn = (c1 / dz) / img_wh[cam * 2 + 1];
  bool ok = (depth > 1e-5f) && (un > 0.0f) && (un < 1.0f) && (vn > 0.0f) && (vn < 1.0f);
  bx[cam * QTOT + q]  = un;
  by[cam * QTOT + q]  = vn;
  vis[cam * QTOT + q] = ok ? 1.0f : 0.0f;
}

// ---------------------------------------------------------------------------
// 3) value transpose: ref[L,Ce,Hl,Wl] -> val[L,Hl,Wl,Ce]  (channel-last)
// ---------------------------------------------------------------------------
__global__ void tval_kernel(const float* __restrict__ ref, float* __restrict__ val) {
  int i = blockIdx.x * blockDim.x + threadIdx.x;
  const int TOT = NCAM * CE * HL * WL;
  if (i >= TOT) return;
  int x = i % WL;
  int y = (i / WL) % HL;
  int c = (i / (WL * HL)) % CE;
  int l = i / (WL * HL * CE);
  val[(((size_t)l * HL + y) * WL + x) * CE + c] = ref[i];
}

// ---------------------------------------------------------------------------
// 4a) LN over channels, input layout [Ce, Q] -> bf16 [Q, Ce]
// ---------------------------------------------------------------------------
__global__ void ln_cols_kernel(const float* __restrict__ x,
                               const float* __restrict__ g,
                               const float* __restrict__ b,
                               unsigned short* __restrict__ outbf) {
  int q = blockIdx.x * blockDim.x + threadIdx.x;
  if (q >= QTOT) return;
  float s = 0.0f, s2 = 0.0f;
  for (int c = 0; c < CE; ++c) {
    float v = x[(size_t)c * QTOT + q];
    s += v; s2 += v * v;
  }
  float mean = s * (1.0f / CE);
  float var  = s2 * (1.0f / CE) - mean * mean;
  float rstd = rsqrtf(var + 1e-6f);
  for (int c = 0; c < CE; ++c) {
    float v = x[(size_t)c * QTOT + q];
    outbf[(size_t)q * CE + c] = f2bf((v - mean) * rstd * g[c] + b[c]);
  }
}

// ---------------------------------------------------------------------------
// 4b) LN over channels, input layout [Q, Ce] (one wave32 per row) -> bf16 [Q,Ce]
// ---------------------------------------------------------------------------
__global__ __launch_bounds__(256)
void ln_rows_kernel(const float* __restrict__ in,
                    const float* __restrict__ g,
                    const float* __restrict__ b,
                    unsigned short* __restrict__ outbf) {
  int wave = (int)((blockIdx.x * blockDim.x + threadIdx.x) >> 5);
  int lane = threadIdx.x & 31;
  if (wave >= QTOT) return;
  const float* row = in + (size_t)wave * CE;
  float v[8];
  float s = 0.0f, s2 = 0.0f;
#pragma unroll
  for (int j = 0; j < 8; ++j) {
    v[j] = row[j * 32 + lane];
    s += v[j]; s2 += v[j] * v[j];
  }
#pragma unroll
  for (int m = 16; m > 0; m >>= 1) {
    s  += __shfl_xor(s,  m, 32);
    s2 += __shfl_xor(s2, m, 32);
  }
  float mean = s * (1.0f / CE);
  float var  = s2 * (1.0f / CE) - mean * mean;
  float rstd = rsqrtf(var + 1e-6f);
#pragma unroll
  for (int j = 0; j < 8; ++j) {
    int c = j * 32 + lane;
    outbf[(size_t)wave * CE + c] = f2bf((v[j] - mean) * rstd * g[c] + b[c]);
  }
}

// ---------------------------------------------------------------------------
// 5) WMMA GEMM: Out[q, o] = sum_c A[q,c] * W[o,c] + bias[o]
//    A: [Q,256] bf16 row-major, W: [NOUT,256] bf16 row-major
//    One wave32 computes a 16x16 tile via 8x v_wmma_f32_16x16x32_bf16.
//    MODE 0: bf16 out [Q,NOUT]   MODE 1: f32 tanh out [Q,NOUT]
//    MODE 2: f32 out [Q,NOUT]    MODE 3: f32 residual out [NOUT_tot? -> [Ce,Q]]
// ---------------------------------------------------------------------------
template <int NOUT, int MODE>
__global__ __launch_bounds__(32)
void gemm_kernel(const unsigned short* __restrict__ A,
                 const unsigned short* __restrict__ W,
                 const float* __restrict__ bias,
                 const float* __restrict__ resid,
                 void* __restrict__ Out) {
  const int lane  = threadIdx.x;
  const int hl    = lane >> 4;       // half: 0 or 1
  const int ln16  = lane & 15;
  const int qbase = blockIdx.x * 16; // Q = 10000 = 625*16 exactly
  const int nbase = blockIdx.y * 16;

  FragC acc;
#pragma unroll
  for (int i = 0; i < 8; ++i) acc.f[i] = 0.0f;

  const unsigned short* arow = A + (size_t)(qbase + ln16) * CE;
  const unsigned short* brow = W + (size_t)(nbase + ln16) * CE;

#pragma unroll
  for (int kb = 0; kb < CE; kb += 32) {
    FragAB a, b;
#pragma unroll
    for (int v = 0; v < 8; ++v) {   // A: lane=M, K pairs per documented layout
      int k0 = (v >> 2) * 16 + hl * 8 + (v & 3) * 2;
      a.d[v] = *(const unsigned int*)(arow + kb + k0);
    }
#pragma unroll
    for (int j = 0; j < 8; ++j) {   // B: lane=N, K pairs same order as A
      b.d[j] = *(const unsigned int*)(brow + kb + (j >> 2) * 16 + hl * 8 + (j & 3) * 2);
    }
    acc.v = __builtin_amdgcn_wmma_f32_16x16x32_bf16(
        false, a.v, false, b.v, (short)0, acc.v, false, false);
  }

#pragma unroll
  for (int r = 0; r < 8; ++r) {
    int m = r + 8 * hl;   // D layout: VGPR r -> row r (lanes 0-15) / r+8 (16-31)
    int n = ln16;
    int q = qbase + m;
    int o = nbase + n;
    float val = acc.f[r] + bias[o];
    if (MODE == 0) {
      ((unsigned short*)Out)[(size_t)q * NOUT + o] = f2bf(val);
    } else if (MODE == 1) {
      ((float*)Out)[(size_t)q * NOUT + o] = tanhf(val);
    } else if (MODE == 2) {
      ((float*)Out)[(size_t)q * NOUT + o] = val;
    } else {
      size_t idx = (size_t)o * QTOT + q;        // output layout [Ce, Q]
      ((float*)Out)[idx] = resid[idx] + val;
    }
  }
}

// ---------------------------------------------------------------------------
// 6) softmax over NP=4 within each (q, head)
// ---------------------------------------------------------------------------
__global__ void softmax_np_kernel(float* __restrict__ att) {
  int i = blockIdx.x * blockDim.x + threadIdx.x;
  if (i >= QTOT * NHEAD) return;
  float* a = att + (size_t)i * NPNT;
  float m = fmaxf(fmaxf(a[0], a[1]), fmaxf(a[2], a[3]));
  float e0 = expf(a[0] - m), e1 = expf(a[1] - m);
  float e2 = expf(a[2] - m), e3 = expf(a[3] - m);
  float inv = 1.0f / (e0 + e1 + e2 + e3);
  a[0] = e0 * inv; a[1] = e1 * inv; a[2] = e2 * inv; a[3] = e3 * inv;
}

// ---------------------------------------------------------------------------
// 7) deformable sampling: one block per q, one wave32 per head (lane = channel)
//    val is channel-last -> each corner gather is a coalesced 128B read (L2 hit)
// ---------------------------------------------------------------------------
__global__ __launch_bounds__(256)
void sampler_kernel(const float* __restrict__ off,   // [Q,64] (tanh applied)
                    const float* __restrict__ att,   // [Q,32] (softmaxed)
                    const float* __restrict__ bx,    // [Nc,Q]
                    const float* __restrict__ by,
                    const float* __restrict__ vis,
                    const float* __restrict__ val,   // [L,Hl,Wl,Ce]
                    float* __restrict__ out) {       // [Q,Ce]
  __shared__ float s_off[NHEAD * NPNT * 2];
  __shared__ float s_att[NHEAD * NPNT];
  __shared__ float s_bx[NCAM], s_by[NCAM], s_vis[NCAM];
  int q = blockIdx.x;
  int tid = threadIdx.x;
  if (tid < 64) {
    s_off[tid] = off[(size_t)q * 64 + tid];
  } else if (tid < 96) {
    s_att[tid - 64] = att[(size_t)q * 32 + (tid - 64)];
  } else if (tid < 96 + NCAM) {
    int l = tid - 96;
    s_bx[l]  = bx[l * QTOT + q];
    s_by[l]  = by[l * QTOT + q];
    s_vis[l] = vis[l * QTOT + q];
  }
  __syncthreads();

  int h = tid >> 5;   // head
  int d = tid & 31;   // channel within head

  float vsum = 0.0f;
#pragma unroll
  for (int l = 0; l < NCAM; ++l) vsum += s_vis[l];
  float asum = s_att[h*4+0] + s_att[h*4+1] + s_att[h*4+2] + s_att[h*4+3];
  float invden = 1.0f / (vsum * asum + 1e-6f);

  float acc = 0.0f;
  for (int l = 0; l < NCAM; ++l) {
    float vl = s_vis[l];
    float bxl = s_bx[l] * (float)WL - 0.5f;
    float byl = s_by[l] * (float)HL - 0.5f;
    const float* vbase = val + (size_t)l * HL * WL * CE + h * DH + d;
#pragma unroll
    for (int p = 0; p < NPNT; ++p) {
      float watt = s_att[h * 4 + p] * vl * invden;
      float xp = bxl + s_off[h * 8 + p * 2 + 0];
      float yp = byl + s_off[h * 8 + p * 2 + 1];
      float x0f = floorf(xp), y0f = floorf(yp);
      int x0 = (int)x0f, y0 = (int)y0f;
      float wx1 = xp - x0f, wy1 = yp - y0f;
      float wx0 = 1.0f - wx1, wy0 = 1.0f - wy1;
      float s = 0.0f;
#pragma unroll
      for (int cy = 0; cy < 2; ++cy) {
#pragma unroll
        for (int cx = 0; cx < 2; ++cx) {
          int ix = x0 + cx, iy = y0 + cy;
          bool okc = (ix >= 0) && (ix < WL) && (iy >= 0) && (iy < HL);
          int cix = min(max(ix, 0), WL - 1);
          int ciy = min(max(iy, 0), HL - 1);
          float g = vbase[((size_t)ciy * WL + cix) * CE];
          float w = (cx ? wx1 : wx0) * (cy ? wy1 : wy0);
          s += okc ? (w * g) : 0.0f;
        }
      }
      acc += watt * s;
    }
  }
  out[(size_t)q * CE + h * DH + d] = acc;
}

// ---------------------------------------------------------------------------
// Launch
// ---------------------------------------------------------------------------
extern "C" void kernel_launch(void* const* d_in, const int* in_sizes, int n_in,
                              void* d_out, int out_size, void* d_ws, size_t ws_size,
                              hipStream_t stream) {
  const float* x        = (const float*)d_in[0];   // [Ce, Q]
  const float* ref      = (const float*)d_in[1];   // [Nc, Ce, Hl, Wl]
  const float* i2l      = (const float*)d_in[2];   // [Nc, 4, 4]
  const float* img_wh   = (const float*)d_in[3];   // [Nc, 2]
  const float* ln_in_g  = (const float*)d_in[4];
  const float* ln_in_b  = (const float*)d_in[5];
  const float* ff_in_w  = (const float*)d_in[6];
  const float* ff_in_b  = (const float*)d_in[7];
  const float* offset_w = (const float*)d_in[8];
  const float* offset_b = (const float*)d_in[9];
  const float* attn_w   = (const float*)d_in[10];
  const float* attn_b   = (const float*)d_in[11];
  const float* ln_out_g = (const float*)d_in[12];
  const float* ln_out_b = (const float*)d_in[13];
  const float* ff_out_w = (const float*)d_in[14];
  const float* ff_out_b = (const float*)d_in[15];
  (void)in_sizes; (void)n_in; (void)out_size; (void)ws_size;

  // workspace carve-up (all offsets 512B-aligned); total ~48 MB
  char* base = (char*)d_ws;
  size_t cur = 0;
  auto take = [&](size_t bytes) -> char* {
    char* r = base + cur;
    cur += (bytes + 511) & ~(size_t)511;
    return r;
  };
  unsigned short* xln  = (unsigned short*)take((size_t)QTOT * CE * 2);
  unsigned short* hbf  = (unsigned short*)take((size_t)QTOT * CE * 2);
  unsigned short* wib  = (unsigned short*)take((size_t)CE * CE * 2);
  unsigned short* wob  = (unsigned short*)take((size_t)NHEAD * NPNT * 2 * CE * 2);
  unsigned short* wab  = (unsigned short*)take((size_t)NHEAD * NPNT * CE * 2);
  unsigned short* wfb  = (unsigned short*)take((size_t)CE * CE * 2);
  float* offb = (float*)take((size_t)QTOT * 64 * 4);
  float* attb = (float*)take((size_t)QTOT * 32 * 4);
  float* bxb  = (float*)take((size_t)NCAM * QTOT * 4);
  float* byb  = (float*)take((size_t)NCAM * QTOT * 4);
  float* visb = (float*)take((size_t)NCAM * QTOT * 4);
  float* valt = (float*)take((size_t)NCAM * HL * WL * CE * 4);
  float* outq = (float*)take((size_t)QTOT * CE * 4);
  unsigned short* oln = (unsigned short*)take((size_t)QTOT * CE * 2);

  // 1) weights -> bf16
  cvt_weights_kernel<<<(CE * CE + 255) / 256, 256, 0, stream>>>(
      ff_in_w, offset_w, attn_w, ff_out_w, wib, wob, wab, wfb);

  // 2) projection (grid.y = camera)
  proj_kernel<<<dim3((QTOT + 255) / 256, NCAM), 256, 0, stream>>>(
      i2l, img_wh, bxb, byb, visb);

  // 3) value transpose to channel-last
  {
    int tot = NCAM * CE * HL * WL;
    tval_kernel<<<(tot + 255) / 256, 256, 0, stream>>>(ref, valt);
  }

  // 4) LN(x) -> bf16 [Q,Ce]
  ln_cols_kernel<<<(QTOT + 255) / 256, 256, 0, stream>>>(x, ln_in_g, ln_in_b, xln);

  // 5) h = xln @ ff_in_w^T + b  (bf16 out)
  gemm_kernel<CE, 0><<<dim3(QTOT / 16, CE / 16), 32, 0, stream>>>(
      xln, wib, ff_in_b, nullptr, (void*)hbf);

  // 6) offsets = tanh(h @ offset_w^T + b)  [Q,64]
  gemm_kernel<64, 1><<<dim3(QTOT / 16, 64 / 16), 32, 0, stream>>>(
      hbf, wob, offset_b, nullptr, (void*)offb);

  // 7) attention logits = h @ attn_w^T + b  [Q,32], then softmax over NP
  gemm_kernel<32, 2><<<dim3(QTOT / 16, 32 / 16), 32, 0, stream>>>(
      hbf, wab, attn_b, nullptr, (void*)attb);
  softmax_np_kernel<<<(QTOT * NHEAD + 255) / 256, 256, 0, stream>>>(attb);

  // 8) deformable sampling -> out [Q,Ce]
  sampler_kernel<<<QTOT, 256, 0, stream>>>(offb, attb, bxb, byb, visb, valt, outq);

  // 9) LN(out) -> bf16 [Q,Ce]
  ln_rows_kernel<<<(QTOT * 32 + 255) / 256, 256, 0, stream>>>(
      outq, ln_out_g, ln_out_b, oln);

  // 10) y = oln @ ff_out_w^T + b, written as [Ce,Q] with residual x
  gemm_kernel<CE, 3><<<dim3(QTOT / 16, CE / 16), 32, 0, stream>>>(
      oln, wfb, ff_out_b, x, d_out);
}
